// SimpleRNNClassifier_46377056862584
// MI455X (gfx1250) — matrix operand order repaired and checked by reading, not compiled, not run on hardware
//
#include <hip/hip_runtime.h>
#include <hip/hip_bf16.h>

// ---------------- problem constants ----------------
constexpr int Vv = 32000, E_ = 256, H_ = 512, B_ = 256, S_ = 512, C_ = 2;

// ---------------- vector types ----------------
typedef __bf16 v16bf __attribute__((ext_vector_type(16)));
typedef __bf16 v8bf  __attribute__((ext_vector_type(8)));
typedef float  v8f   __attribute__((ext_vector_type(8)));

__device__ __forceinline__ v8bf ld8(const __bf16* p) {
    return *reinterpret_cast<const v8bf*>(p);   // 16B aligned -> global_load_b128
}
__device__ __forceinline__ v16bf cat(v8bf lo, v8bf hi) {
    return __builtin_shufflevector(lo, hi, 0,1,2,3,4,5,6,7,8,9,10,11,12,13,14,15);
}

// ---------------- workspace layout (bytes) ----------------
constexpr size_t al256(size_t x) { return (x + 255) & ~(size_t)255; }
constexpr size_t OFF_BAR   = 0;                                   // 2 uints (barrier cnt, gen)
constexpr size_t OFF_EMBB  = 256;                                 // [V,E] bf16
constexpr size_t OFF_WXH0T = al256(OFF_EMBB  + (size_t)Vv*E_*2);  // [H,E] bf16 (transposed)
constexpr size_t OFF_WHH0T = al256(OFF_WXH0T + (size_t)H_*E_*2);  // [H,H] bf16
constexpr size_t OFF_WXH1T = al256(OFF_WHH0T + (size_t)H_*H_*2);  // [H,H] bf16
constexpr size_t OFF_WHH1T = al256(OFF_WXH1T + (size_t)H_*H_*2);  // [H,H] bf16
constexpr size_t OFF_H0    = al256(OFF_WHH1T + (size_t)H_*H_*2);  // [2,B,H] bf16 ping/pong
constexpr size_t OFF_H1    = al256(OFF_H0    + (size_t)2*B_*H_*2);// [2,B,H] bf16
constexpr size_t OFF_PRE0  = al256(OFF_H1    + (size_t)2*B_*H_*2);// [S*B,H] f32

// ---------------- grid barrier (persistent kernel, agent scope) ----------------
__device__ __forceinline__ void gridBarrier(unsigned* cnt, unsigned* gen, unsigned nblk) {
    __syncthreads();
    if (threadIdx.x == 0) {
        unsigned g = __hip_atomic_load(gen, __ATOMIC_RELAXED, __HIP_MEMORY_SCOPE_AGENT);
        unsigned a = __hip_atomic_fetch_add(cnt, 1u, __ATOMIC_ACQ_REL, __HIP_MEMORY_SCOPE_AGENT);
        if (a == nblk - 1u) {
            __hip_atomic_store(cnt, 0u, __ATOMIC_RELAXED, __HIP_MEMORY_SCOPE_AGENT);
            __hip_atomic_store(gen, g + 1u, __ATOMIC_RELEASE, __HIP_MEMORY_SCOPE_AGENT);
        } else {
            while (__hip_atomic_load(gen, __ATOMIC_ACQUIRE, __HIP_MEMORY_SCOPE_AGENT) == g)
                __builtin_amdgcn_s_sleep(2);
        }
    }
    __syncthreads();
}

// ---------------- prep kernels ----------------
__global__ void init_kernel(unsigned* bar, __bf16* h0, __bf16* h1) {
    int i = blockIdx.x * blockDim.x + threadIdx.x;
    if (i < 2) bar[i] = 0u;
    int total = 2 * B_ * H_;
    for (int j = i; j < total; j += gridDim.x * blockDim.x) {
        h0[j] = (__bf16)0.0f;
        h1[j] = (__bf16)0.0f;
    }
}

__global__ void cvt_kernel(const float* __restrict__ src, __bf16* __restrict__ dst, int n) {
    int i = blockIdx.x * blockDim.x + threadIdx.x;
    for (int j = i; j < n; j += gridDim.x * blockDim.x) dst[j] = (__bf16)src[j];
}

// dst[n*K + k] = src[k*N + n]   (src is [K x N] f32, dst is [N x K] bf16)
__global__ void tcvt_kernel(const float* __restrict__ src, __bf16* __restrict__ dst, int K, int N) {
    int i = blockIdx.x * blockDim.x + threadIdx.x;
    int total = K * N;
    for (int j = i; j < total; j += gridDim.x * blockDim.x) {
        int n = j / K, k = j - n * K;
        dst[j] = (__bf16)src[(size_t)k * N + n];
    }
}

// ---------------- phase 1: pre0[t*B+b, :] = emb[x[b,t]] @ Wxh0 + bh0 ----------------
__global__ void __launch_bounds__(256) pre0_gemm_kernel(
    const int*    __restrict__ x,      // [B,S]
    const __bf16* __restrict__ embB,   // [V,E] bf16
    const __bf16* __restrict__ wxh0T,  // [H,E] bf16 (transposed)
    const float*  __restrict__ bh0,    // [H]
    float*        __restrict__ pre0)   // [S*B, H]
{
    const int lane = threadIdx.x & 31;
    const int wave = blockIdx.x * 8 + (threadIdx.x >> 5);
    const int nT = wave & 31;          // H/16 = 32 tiles
    const int mT = wave >> 5;          // (S*B)/16 = 8192 tiles
    const int m0 = mT * 16, n0 = nT * 16;
    const int ln = lane & 15, lh = lane >> 4;

    // gather: this lane's A-row token
    const int fA  = m0 + ln;           // flat row index = t*B + b
    const int bA  = fA & (B_ - 1);
    const int tA  = fA >> 8;           // B_ == 256
    const int tok = x[bA * S_ + tA];

    const __bf16* aRow = embB  + (size_t)tok * E_;
    const __bf16* bRow = wxh0T + (size_t)(n0 + ln) * E_;

    v8f acc;
    const float bv = bh0[n0 + ln];
    #pragma unroll
    for (int r = 0; r < 8; ++r) acc[r] = bv;

    #pragma unroll
    for (int k = 0; k < E_; k += 32) {
        v16bf A  = cat(ld8(aRow + k + lh * 8),  ld8(aRow + k + 16 + lh * 8));
        v16bf Bm = cat(ld8(bRow + k + lh * 8), ld8(bRow + k + 16 + lh * 8));
        acc = __builtin_amdgcn_wmma_f32_16x16x32_bf16(false, A, false, Bm,
                                                      (short)0, acc, false, false);
    }

    float* out = pre0 + (size_t)m0 * H_ + n0 + ln;
    #pragma unroll
    for (int r = 0; r < 8; ++r) out[(size_t)(r + 8 * lh) * H_] = acc[r];
}

// ---------------- phase 2: persistent sequential scan ----------------
// One phase per step computes BOTH h1(t) and h0n(t+1) from h0n(t)/h1(t-1):
//   h1(t)     = tanh(h0n(t) @ Wxh1 + h1(t-1) @ Whh1 + bh1)
//   h0n(t+1)  = tanh(pre0[t+1] + h0n(t) @ Whh0)
// => exactly ONE grid barrier per step. All three weight column-blocks are
// register-resident (3 * 16 fragments * 8 VGPRs = 384 VGPRs) for the whole scan.
__global__ void __launch_bounds__(256, 1) rnn_scan_kernel(
    const float*  __restrict__ pre0,   // [S*B, H]
    const __bf16* __restrict__ whh0T,  // [H,H]
    const __bf16* __restrict__ wxh1T,  // [H,H]
    const __bf16* __restrict__ whh1T,  // [H,H]
    const float*  __restrict__ bh1,    // [H]
    __bf16*       __restrict__ h0buf,  // [2,B,H]
    __bf16*       __restrict__ h1buf,  // [2,B,H]
    unsigned*     __restrict__ bar)
{
    const int lane = threadIdx.x & 31;
    const int wave = blockIdx.x * 8 + (threadIdx.x >> 5); // 64 blk * 8 = 512 waves
    const int nT = wave & 31, mT = wave >> 5;             // 32 x 16 tiles
    const int m0 = mT * 16, n0 = nT * 16;
    const int ln = lane & 15, lh = lane >> 4;
    const unsigned NBLK = gridDim.x;
    unsigned* cnt = bar;
    unsigned* gen = bar + 1;
    constexpr size_t HB = (size_t)B_ * H_;

    // ---- hoist all three weight column-blocks into registers ----
    v16bf w0[16], w1[16], w2[16];          // Whh0, Wxh1, Whh1
    {
        const __bf16* r0 = whh0T + (size_t)(n0 + ln) * H_ + lh * 16;
        const __bf16* r1 = wxh1T + (size_t)(n0 + ln) * H_ + lh * 16;
        const __bf16* r2 = whh1T + (size_t)(n0 + ln) * H_ + lh * 16;
        #pragma unroll
        for (int kk = 0; kk < 16; ++kk) {
            w0[kk] = cat(ld8(r0 + kk * 32), ld8(r0 + kk * 32 + 8));
            w1[kk] = cat(ld8(r1 + kk * 32), ld8(r1 + kk * 32 + 8));
            w2[kk] = cat(ld8(r2 + kk * 32), ld8(r2 + kk * 32 + 8));
        }
    }
    const float bv1 = bh1[n0 + ln];

    // ---- prologue: h0n(0) = tanh(pre0[0])  (h0(-1) == 0) ----
    {
        const float* pr = pre0 + (size_t)m0 * H_ + n0 + ln;
        __bf16* out = h0buf + (size_t)m0 * H_ + n0 + ln;   // buffer 0
        #pragma unroll
        for (int r = 0; r < 8; ++r)
            out[(size_t)(r + 8 * lh) * H_] = (__bf16)tanhf(pr[(size_t)(r + 8 * lh) * H_]);
    }
    gridBarrier(cnt, gen, NBLK);

    // ---- main loop: one barrier per step ----
    for (int t = 0; t < S_; ++t) {
        const __bf16* h0cur = h0buf + (size_t)(t & 1) * HB;        // h0n(t)
        __bf16*       h0nxt = h0buf + (size_t)((t + 1) & 1) * HB;  // h0n(t+1)
        const __bf16* h1prv = h1buf + (size_t)(t & 1) * HB;        // h1(t-1)
        __bf16*       h1cur = h1buf + (size_t)((t + 1) & 1) * HB;  // h1(t)
        const bool hasNext = (t + 1 < S_);
        const int  tn      = hasNext ? (t + 1) : t;

        v8f acc1;                                   // -> h1(t)
        #pragma unroll
        for (int r = 0; r < 8; ++r) acc1[r] = bv1;

        v8f acc0;                                   // -> h0n(t+1)
        {
            const float* pr = pre0 + ((size_t)tn * B_ + m0) * H_ + n0 + ln;
            #pragma unroll
            for (int r = 0; r < 8; ++r) acc0[r] = pr[(size_t)(r + 8 * lh) * H_];
        }

        const __bf16* aRowC = h0cur + (size_t)(m0 + ln) * H_;   // shared A: h0n(t)
        const __bf16* aRowP = h1prv + (size_t)(m0 + ln) * H_;   // A: h1(t-1)
        #pragma unroll
        for (int kk = 0; kk < 16; ++kk) {
            const int k = kk * 32;
            v16bf A = cat(ld8(aRowC + k + lh * 16), ld8(aRowC + k + lh * 16 + 8));
            acc1 = __builtin_amdgcn_wmma_f32_16x16x32_bf16(false, A, false, w1[kk],
                                                           (short)0, acc1, false, false);
            acc0 = __builtin_amdgcn_wmma_f32_16x16x32_bf16(false, A, false, w0[kk],
                                                           (short)0, acc0, false, false);
            v16bf A2 = cat(ld8(aRowP + k + lh * 16), ld8(aRowP + k + lh * 16 + 8));
            acc1 = __builtin_amdgcn_wmma_f32_16x16x32_bf16(false, A2, false, w2[kk],
                                                           (short)0, acc1, false, false);
        }

        {
            __bf16* out = h1cur + (size_t)m0 * H_ + n0 + ln;
            #pragma unroll
            for (int r = 0; r < 8; ++r)
                out[(size_t)(r + 8 * lh) * H_] = (__bf16)tanhf(acc1[r]);
        }
        if (hasNext) {
            __bf16* out = h0nxt + (size_t)m0 * H_ + n0 + ln;
            #pragma unroll
            for (int r = 0; r < 8; ++r)
                out[(size_t)(r + 8 * lh) * H_] = (__bf16)tanhf(acc0[r]);
        }
        gridBarrier(cnt, gen, NBLK);
    }
}

// ---------------- head: out = h1 @ Wc + bc  (256x512 @ 512x2) ----------------
__global__ void head_kernel(const __bf16* __restrict__ h1,   // [B,H] (final ping buffer)
                            const float*  __restrict__ Wc,   // [H,C]
                            const float*  __restrict__ bc,   // [C]
                            float*        __restrict__ out)  // [B,C]
{
    int i = blockIdx.x * blockDim.x + threadIdx.x;
    if (i >= B_ * C_) return;
    int b = i >> 1, c = i & 1;
    float acc = bc[c];
    const __bf16* row = h1 + (size_t)b * H_;
    #pragma unroll 8
    for (int h = 0; h < H_; ++h) acc += (float)row[h] * Wc[h * C_ + c];
    out[i] = acc;
}

// ---------------- launcher ----------------
extern "C" void kernel_launch(void* const* d_in, const int* in_sizes, int n_in,
                              void* d_out, int out_size, void* d_ws, size_t ws_size,
                              hipStream_t stream) {
    const int*   x    = (const int*)  d_in[0];
    const float* emb  = (const float*)d_in[1];
    const float* Wxh0 = (const float*)d_in[2];
    const float* Whh0 = (const float*)d_in[3];
    const float* bh0  = (const float*)d_in[4];
    const float* Wxh1 = (const float*)d_in[5];
    const float* Whh1 = (const float*)d_in[6];
    const float* bh1  = (const float*)d_in[7];
    const float* Wc   = (const float*)d_in[8];
    const float* bc   = (const float*)d_in[9];
    float* out = (float*)d_out;

    char* ws = (char*)d_ws;
    unsigned* bar    = (unsigned*)(ws + OFF_BAR);
    __bf16*   embB   = (__bf16*)  (ws + OFF_EMBB);
    __bf16*   wxh0T  = (__bf16*)  (ws + OFF_WXH0T);
    __bf16*   whh0T  = (__bf16*)  (ws + OFF_WHH0T);
    __bf16*   wxh1T  = (__bf16*)  (ws + OFF_WXH1T);
    __bf16*   whh1T  = (__bf16*)  (ws + OFF_WHH1T);
    __bf16*   h0buf  = (__bf16*)  (ws + OFF_H0);
    __bf16*   h1buf  = (__bf16*)  (ws + OFF_H1);
    float*    pre0   = (float*)   (ws + OFF_PRE0);

    // phase 0: init state + convert/transposed weights to bf16
    init_kernel<<<256, 256, 0, stream>>>(bar, h0buf, h1buf);
    cvt_kernel <<<1024, 256, 0, stream>>>(emb, embB, Vv * E_);
    tcvt_kernel<<<512, 256, 0, stream>>>(Wxh0, wxh0T, E_, H_);   // -> [H,E]
    tcvt_kernel<<<512, 256, 0, stream>>>(Whh0, whh0T, H_, H_);
    tcvt_kernel<<<512, 256, 0, stream>>>(Wxh1, wxh1T, H_, H_);
    tcvt_kernel<<<512, 256, 0, stream>>>(Whh1, whh1T, H_, H_);

    // phase 1: parallel input projection (131072 x 256 @ 256 x 512)
    pre0_gemm_kernel<<<32768, 256, 0, stream>>>(x, embB, wxh0T, bh0, pre0);

    // phase 2: persistent sequential scan (64 blocks * 8 waves = 512 tile-waves)
    rnn_scan_kernel<<<64, 256, 0, stream>>>(pre0, whh0T, wxh1T, whh1T, bh1,
                                            h0buf, h1buf, bar);

    // head: final h1 lives in ping buffer 0 (h1(S-1) written at t=S-1 -> (S&1)==0)
    head_kernel<<<2, 256, 0, stream>>>(h1buf, Wc, bc, out);
}
